// TemporalAttentionLayer_10608569221673
// MI455X (gfx1250) — matrix operand order repaired and checked by reading, not compiled, not run on hardware
//
#include <hip/hip_runtime.h>

// CDNA5 / gfx1250 wave32 WMMA types
typedef __attribute__((ext_vector_type(16))) _Float16 v16h;
typedef __attribute__((ext_vector_type(8)))  float    v8f;

static constexpr int T_  = 64;    // time steps
static constexpr int D_  = 256;   // model dim
static constexpr int SCS = 72;    // padded score-row stride (f16 elems)

#define WMMA_F16(A, B, C) \
  __builtin_amdgcn_wmma_f32_16x16x32_f16(false, (A), false, (B), (short)0, (C), false, false)

// A fragment (16x32 f16, MxK): lane (half,m) holds k = 16*half + i
__device__ __forceinline__ v16h ld_a16(const _Float16* p, int row0, int ld,
                                       int kbase, int kreal) {
  const int lane = threadIdx.x & 31, lr = lane & 15, lh = lane >> 4;
  v16h a;
#pragma unroll
  for (int i = 0; i < 16; ++i) {
    const int k = lh * 16 + i;
    a[i] = (k < kreal) ? p[(row0 + lr) * ld + kbase + k] : (_Float16)0.0f;
  }
  return a;
}

// B fragment (32x16 f16, KxN): lane (half,n) holds k = 16*half + i ; K along rows of p
__device__ __forceinline__ v16h ld_b16(const _Float16* p, int k0, int ld,
                                       int col0, int kreal) {
  const int lane = threadIdx.x & 31, lr = lane & 15, lh = lane >> 4;
  v16h b;
#pragma unroll
  for (int i = 0; i < 16; ++i) {
    const int k = lh * 16 + i;
    b[i] = (k < kreal) ? p[(k0 + k) * ld + col0 + lr] : (_Float16)0.0f;
  }
  return b;
}

// B fragment where B = M^T of a row-major f16 matrix (K along columns of p)
__device__ __forceinline__ v16h ld_bT16(const _Float16* p, int row0, int ld,
                                        int colbase, int kreal) {
  const int lane = threadIdx.x & 31, lr = lane & 15, lh = lane >> 4;
  v16h b;
#pragma unroll
  for (int i = 0; i < 16; ++i) {
    const int k = lh * 16 + i;
    b[i] = (k < kreal) ? p[(row0 + lr) * ld + colbase + k] : (_Float16)0.0f;
  }
  return b;
}

// B fragment from f32 global weights (row-major [K][N]), convert to f16
__device__ __forceinline__ v16h ld_b32(const float* p, int k0, int ld, int col0) {
  const int lane = threadIdx.x & 31, lr = lane & 15, lh = lane >> 4;
  v16h b;
#pragma unroll
  for (int i = 0; i < 16; ++i) {
    const int k = lh * 16 + i;
    b[i] = (_Float16)p[(k0 + k) * ld + col0 + lr];
  }
  return b;
}

__global__ __launch_bounds__(256)
void temporal_attention_kernel(const float* __restrict__ x,
                               const float* __restrict__ pos,
                               const float* __restrict__ Wq,
                               const float* __restrict__ Wk,
                               const float* __restrict__ Wv,
                               const float* __restrict__ Wff,
                               const float* __restrict__ bff,
                               float* __restrict__ out) {
  __shared__ alignas(16) _Float16 h16[T_ * D_];   // phase1/2: h ; phase3/4: attn out
  __shared__ alignas(16) _Float16 q16[T_ * D_];
  __shared__ alignas(16) _Float16 k16[T_ * D_];
  __shared__ alignas(16) _Float16 v16s[T_ * D_];
  __shared__ alignas(16) _Float16 sc[8][T_ * SCS]; // scores; phase1: raw-x staging

  const int tid  = threadIdx.x;
  const int wave = tid >> 5;
  const int lane = tid & 31;
  const int lr   = lane & 15;
  const int lh   = lane >> 4;
  const size_t nodeBase = (size_t)blockIdx.x * (T_ * D_);

  // ---------------- phase 1: h = x + pos_emb  (f32 -> f16 in LDS) -------------
  // Stage raw x through LDS with gfx1250 async DMA loads (no VGPR round trip).
  {
    float* xstage = (float*)&sc[0][0];           // 64KB staging inside 72KB region
    const float* xg = x + nodeBase;
#pragma unroll 1
    for (int i = tid; i < T_ * D_ / 4; i += 256) {
      const unsigned ldsOff = (unsigned)(size_t)(xstage + 4 * i);
      const float* gaddr    = xg + 4 * i;
      asm volatile("global_load_async_to_lds_b128 %0, %1, off"
                   :: "v"(ldsOff), "v"(gaddr) : "memory");
    }
    asm volatile("s_wait_asynccnt 0" ::: "memory");
    __syncthreads();
    const float4* xv = (const float4*)xstage;
    const float4* pv = (const float4*)pos;
#pragma unroll 1
    for (int i = tid; i < T_ * D_ / 4; i += 256) {
      const float4 a = xv[i];
      const float4 b = pv[i];
      h16[4 * i + 0] = (_Float16)(a.x + b.x);
      h16[4 * i + 1] = (_Float16)(a.y + b.y);
      h16[4 * i + 2] = (_Float16)(a.z + b.z);
      h16[4 * i + 3] = (_Float16)(a.w + b.w);
    }
  }
  __syncthreads();

  // ---------------- phase 2: q,k,v = h @ W{q,k,v}  (WMMA f16) -----------------
#pragma unroll 1
  for (int jj = 0; jj < 6; ++jj) {
    const int job = wave * 6 + jj;          // 48 jobs: (matrix, col-tile)
    const int mat = job >> 4;
    const int nt  = job & 15;
    const float* W = (mat == 0) ? Wq : (mat == 1) ? Wk : Wv;
    _Float16* dst  = (mat == 0) ? q16 : (mat == 1) ? k16 : v16s;
    v8f acc[4] = {};
#pragma unroll 1
    for (int k0 = 0; k0 < D_; k0 += 32) {
      __builtin_prefetch(W + (size_t)(k0 + 32) * D_ + nt * 16, 0, 0); // speculative
      const v16h bf = ld_b32(W, k0, D_, nt * 16);
#pragma unroll
      for (int mt = 0; mt < 4; ++mt) {
        const v16h af = ld_a16(h16, mt * 16, D_, k0, 32);
        acc[mt] = WMMA_F16(af, bf, acc[mt]);
      }
    }
#pragma unroll
    for (int mt = 0; mt < 4; ++mt)
#pragma unroll
      for (int r = 0; r < 8; ++r)
        dst[(mt * 16 + r + lh * 8) * D_ + nt * 16 + lr] = (_Float16)acc[mt][r];
  }
  __syncthreads();

  // ---------------- phase 3: per-head causal attention ------------------------
  _Float16* ao16 = h16;        // h region reused for attention output (f16)
  _Float16* scb  = sc[wave];
#pragma unroll 1
  for (int hi = 0; hi < 2; ++hi) {
    const int hc = (wave + hi * 8) * 16;   // head column base
    // scores = (q_h @ k_h^T) * (1/sqrt(T)) with causal mask (K=16 padded to 32)
#pragma unroll 1
    for (int st = 0; st < 4; ++st) {
      const v16h bf = ld_bT16(k16, st * 16, D_, hc, 16);
#pragma unroll
      for (int tt = 0; tt < 4; ++tt) {
        const v16h af = ld_a16(q16, tt * 16, D_, hc, 16);
        v8f a = {};
        a = WMMA_F16(af, bf, a);
#pragma unroll
        for (int r = 0; r < 8; ++r) {
          const int t = tt * 16 + r + lh * 8;
          const int s = st * 16 + lr;
          const float v = (s <= t) ? a[r] * 0.125f : -1.0e9f;  // -> -inf in f16
          scb[t * SCS + s] = (_Float16)v;
        }
      }
    }
    __syncthreads();
    // softmax over s: each lane owns rows lane and lane+32
#pragma unroll 1
    for (int rr = 0; rr < 2; ++rr) {
      _Float16* row = scb + (lane + rr * 32) * SCS;
      float mx = -3.0e38f;
#pragma unroll 8
      for (int s = 0; s < T_; ++s) mx = fmaxf(mx, (float)row[s]);
      float sum = 0.0f;
#pragma unroll 8
      for (int s = 0; s < T_; ++s) {
        const float e = __expf((float)row[s] - mx);
        sum += e;
        row[s] = (_Float16)e;
      }
      const float inv = 1.0f / sum;
#pragma unroll 8
      for (int s = 0; s < T_; ++s) row[s] = (_Float16)((float)row[s] * inv);
    }
    __syncthreads();
    // out_head = P @ v_head   (K = 64 in two WMMA steps)
#pragma unroll 1
    for (int tt = 0; tt < 4; ++tt) {
      v8f acc = {};
#pragma unroll
      for (int k0 = 0; k0 < T_; k0 += 32) {
        const v16h af = ld_a16(scb, tt * 16, SCS, k0, 32);
        const v16h bf = ld_b16(v16s, k0, D_, hc, 32);
        acc = WMMA_F16(af, bf, acc);
      }
#pragma unroll
      for (int r = 0; r < 8; ++r)
        ao16[(tt * 16 + r + lh * 8) * D_ + hc + lr] = (_Float16)acc[r];
    }
  }
  __syncthreads();

  // ---------------- phase 4: FFN  y = relu(ao @ Wff + b) + ao -----------------
#pragma unroll 1
  for (int nn = 0; nn < 2; ++nn) {
    const int c0   = (wave * 2 + nn) * 16;
    const float bias = bff[c0 + lr];
    v8f acc[4] = {};
#pragma unroll 1
    for (int k0 = 0; k0 < D_; k0 += 32) {
      __builtin_prefetch(Wff + (size_t)(k0 + 32) * D_ + c0, 0, 0);  // speculative
      const v16h bf = ld_b32(Wff, k0, D_, c0);
#pragma unroll
      for (int mt = 0; mt < 4; ++mt) {
        const v16h af = ld_a16(ao16, mt * 16, D_, k0, 32);
        acc[mt] = WMMA_F16(af, bf, acc[mt]);
      }
    }
#pragma unroll
    for (int mt = 0; mt < 4; ++mt)
#pragma unroll
      for (int r = 0; r < 8; ++r) {
        const int t   = mt * 16 + r + lh * 8;
        const float f = acc[mt][r] + bias;
        const float res = (float)ao16[t * D_ + c0 + lr];
        out[nodeBase + (size_t)(t * D_ + c0 + lr)] = fmaxf(f, 0.0f) + res;
      }
  }
}

extern "C" void kernel_launch(void* const* d_in, const int* in_sizes, int n_in,
                              void* d_out, int out_size, void* d_ws, size_t ws_size,
                              hipStream_t stream) {
  (void)n_in; (void)out_size; (void)d_ws; (void)ws_size;
  const float* x    = (const float*)d_in[0];
  const float* pos  = (const float*)d_in[1];
  const float* Wq   = (const float*)d_in[2];
  const float* Wk   = (const float*)d_in[3];
  const float* Wv   = (const float*)d_in[4];
  const float* Wff  = (const float*)d_in[5];
  const float* bff  = (const float*)d_in[6];
  float* out        = (float*)d_out;

  const int nNodes = in_sizes[0] / (T_ * D_);   // 4096
  hipLaunchKernelGGL(temporal_attention_kernel, dim3(nNodes), dim3(256), 0, stream,
                     x, pos, Wq, Wk, Wv, Wff, bff, out);
}
